// Self_attention_71236327572069
// MI455X (gfx1250) — matrix operand order repaired and checked by reading, not compiled, not run on hardware
//
#include <hip/hip_runtime.h>
#include <hip/hip_bf16.h>

#define TSEQ 2048
#define DIM 512
#define SPLITS 16
#define ROWS_PER_SPLIT (TSEQ / SPLITS)   // 128 key rows per split
#define RBLK 32                          // 32-row M-block: 2 WMMAs per B-fragment
#define NBLOCKS (ROWS_PER_SPLIT / RBLK)  // 4 row-blocks per split

typedef __attribute__((ext_vector_type(16))) __bf16 v16bf;
typedef __attribute__((ext_vector_type(8)))  float  v8f;

union FragBF { v16bf v; uint4 q[2]; };

__device__ __forceinline__ unsigned short f2bf(float f) {
  unsigned u = __float_as_uint(f);
  u += 0x7FFFu + ((u >> 16) & 1u);   // round-to-nearest-even
  return (unsigned short)(u >> 16);
}

// ---------------------------------------------------------------------------
// Convert an f32 weight matrix to bf16 (stays resident in 192MB L2 afterwards)
// ---------------------------------------------------------------------------
__global__ void cvt_bf16_kernel(const float* __restrict__ src,
                                unsigned short* __restrict__ dst, int n4) {
  int i = blockIdx.x * blockDim.x + threadIdx.x;
  if (i < n4) {
    float4 f = ((const float4*)src)[i];
    ushort4 h;
    h.x = f2bf(f.x); h.y = f2bf(f.y); h.z = f2bf(f.z); h.w = f2bf(f.w);
    ((ushort4*)dst)[i] = h;
  }
}

// ---------------------------------------------------------------------------
// q[b,h,:] = X[b,h,T-1,:] @ Wq^T + bq   (f32 for score precision; tiny work)
// ---------------------------------------------------------------------------
__global__ void qproj_kernel(const float* __restrict__ X,
                             const float* __restrict__ Wq,
                             const float* __restrict__ bq,
                             float* __restrict__ qout) {
  const int bh  = blockIdx.x;
  const int tid = threadIdx.x;
  const float* x = X + ((size_t)bh * TSEQ + (TSEQ - 1)) * DIM;
  for (int n = tid; n < DIM; n += 256) {
    const float* wr = Wq + (size_t)n * DIM;
    float s = bq[n];
    for (int k = 0; k < DIM; k += 4) {
      float4 xv = *(const float4*)(x + k);
      float4 wv = *(const float4*)(wr + k);
      s += xv.x * wv.x + xv.y * wv.y + xv.z * wv.z + xv.w * wv.w;
    }
    qout[(size_t)bh * DIM + n] = s;
  }
}

// ---------------------------------------------------------------------------
// Fused K/V projection (bf16 WMMA) + online softmax for the last query row.
// Grid: (B*H) * SPLITS blocks, 256 threads (8 wave32).
// Each 32-row key block: each wave owns 8 N-tiles (waves 0-3 K-proj,
// waves 4-7 V-proj); each B-fragment feeds TWO wmmas (M rows 0-15 / 16-31),
// halving L2 weight traffic (32 flops per L2 byte).
// ---------------------------------------------------------------------------
__global__ __launch_bounds__(256) void attn_kv_kernel(
    const float* __restrict__ X,
    const unsigned short* __restrict__ WkB,
    const unsigned short* __restrict__ WvB,
    const float* __restrict__ Wk_b,
    const float* __restrict__ Wv_b,
    const float* __restrict__ qglob,
    float* __restrict__ partials) {
  __shared__ unsigned short Xb[RBLK * DIM];   // 32 KB  bf16 A-block
  __shared__ float Klds[RBLK * DIM];          // 64 KB  K rows (f32)
  __shared__ float Vlds[RBLK * DIM];          // 64 KB  V rows (f32)
  __shared__ float qs[DIM];                   //  2 KB
  __shared__ float accs[DIM];                 //  2 KB  running numerator
  __shared__ float spart[256];
  __shared__ float sc[RBLK];
  __shared__ float pr[RBLK];
  __shared__ float sM, sL, sMnew, sScale;

  const int tid  = threadIdx.x;
  const int wave = tid >> 5;
  const int lane = tid & 31;
  const int lm   = lane & 15;   // M (A) or N (B/D) index
  const int lh   = lane >> 4;   // half-wave selector

  const int bh    = blockIdx.x / SPLITS;
  const int split = blockIdx.x % SPLITS;

  const float* Xbh = X + (size_t)bh * TSEQ * DIM;

  // wave-uniform tile setup, hoisted out of the row-block loop
  const bool isV = wave >= 4;                       // waves 0-3: K, 4-7: V
  const unsigned short* Wmat = isV ? WvB : WkB;
  const float*          bvec = isV ? Wv_b : Wk_b;
  float bias_r[8];
#pragma unroll
  for (int t = 0; t < 8; ++t)
    bias_r[t] = bvec[((wave & 3) * 8 + t) * 16 + lm];

  qs[tid]         = qglob[bh * DIM + tid];
  qs[tid + 256]   = qglob[bh * DIM + 256 + tid];
  accs[tid]       = 0.0f;
  accs[tid + 256] = 0.0f;
  if (tid == 0) { sM = -1e30f; sL = 0.0f; }
  __syncthreads();

  const int row0 = split * ROWS_PER_SPLIT;

  for (int blk = 0; blk < NBLOCKS; ++blk) {
    const int r0 = row0 + blk * RBLK;

    // ---- Phase 1: stream X block (f32) -> bf16 in LDS -------------------
    {
      const float4* src = (const float4*)(Xbh + (size_t)r0 * DIM);
#pragma unroll
      for (int it = 0; it < (RBLK * DIM / 4) / 256; ++it) {
        int i = it * 256 + tid;
        float4 f = src[i];
        ushort4 h;
        h.x = f2bf(f.x); h.y = f2bf(f.y); h.z = f2bf(f.z); h.w = f2bf(f.w);
        *(ushort4*)&Xb[i * 4] = h;
      }
      if (blk + 1 < NBLOCKS)
        __builtin_prefetch(Xbh + (size_t)(r0 + RBLK) * DIM + tid * 64, 0, 0);
    }
    __syncthreads();

    // ---- Phase 2: K/V projection via v_wmma_f32_16x16x32_bf16 -----------
#pragma unroll 1
    for (int t = 0; t < 8; ++t) {
      const int nt = (wave & 3) * 8 + t;   // N-tile index (0..31)
      const int n  = nt * 16 + lm;
      const float bv = bias_r[t];          // bias folded into C
      v8f c0 = {bv, bv, bv, bv, bv, bv, bv, bv};   // M rows  0..15
      v8f c1 = {bv, bv, bv, bv, bv, bv, bv, bv};   // M rows 16..31
      const unsigned short* wrow  = Wmat + (size_t)n * DIM;
      const unsigned short* arow0 = &Xb[lm * DIM];
      const unsigned short* arow1 = &Xb[(16 + lm) * DIM];
#pragma unroll
      for (int ks = 0; ks < 16; ++ks) {
        const int k0 = ks * 32;
        FragBF a0, a1, b;
        // B (32x16 bf16): halves 0-7 = K[k0+lh*8 ..], halves 8-15 = +16 of column n
        b.q[0] = *(const uint4*)(wrow + k0 + lh * 8);
        b.q[1] = *(const uint4*)(wrow + k0 + 16 + lh * 8);
        // A (16x32 bf16): halves 0-7 = K[k0+lh*8 ..], halves 8-15 = +16
        a0.q[0] = *(const uint4*)(arow0 + k0 + lh * 8);
        a0.q[1] = *(const uint4*)(arow0 + k0 + 16 + lh * 8);
        a1.q[0] = *(const uint4*)(arow1 + k0 + lh * 8);
        a1.q[1] = *(const uint4*)(arow1 + k0 + 16 + lh * 8);
        c0 = __builtin_amdgcn_wmma_f32_16x16x32_bf16(false, a0.v, false, b.v,
                                                     (short)0, c0, false, false);
        c1 = __builtin_amdgcn_wmma_f32_16x16x32_bf16(false, a1.v, false, b.v,
                                                     (short)0, c1, false, false);
      }
      float* outp = isV ? Vlds : Klds;
#pragma unroll
      for (int r = 0; r < 8; ++r) {       // D layout: M = r + 8*lh, N = lm
        outp[(r + 8 * lh) * DIM + n]      = c0[r];
        outp[(16 + r + 8 * lh) * DIM + n] = c1[r];
      }
    }
    __syncthreads();

    // ---- Phase 3: scores s[m] = q . K[m] / sqrt(D) ----------------------
    {
      const int m = tid >> 3, part = tid & 7;    // 32 rows x 8 partials
      const float* kr = &Klds[m * DIM + part * 64];
      const float* qq = &qs[part * 64];
      float s = 0.0f;
#pragma unroll
      for (int j = 0; j < 64; ++j) s += qq[j] * kr[j];
      spart[tid] = s;
    }
    __syncthreads();
    if (tid < RBLK) {
      float s = 0.0f;
#pragma unroll
      for (int p2 = 0; p2 < 8; ++p2) s += spart[tid * 8 + p2];
      sc[tid] = s * 0.04419417382415922f;  // 1/sqrt(512)
    }
    __syncthreads();
    if (tid == 0) {
      float mb = sM;
      for (int i = 0; i < RBLK; ++i) mb = fmaxf(mb, sc[i]);
      sMnew  = mb;
      sScale = __expf(sM - mb);
    }
    __syncthreads();
    if (tid < RBLK) pr[tid] = __expf(sc[tid] - sMnew);
    __syncthreads();
    if (tid == 0) {
      float ps = 0.0f;
      for (int i = 0; i < RBLK; ++i) ps += pr[i];
      sL = sL * sScale + ps;
      sM = sMnew;
    }
    // ---- Phase 4: online accumulate acc = acc*scale + sum_i p[i]*V[i] ---
    const float es = sScale;
    for (int d = tid; d < DIM; d += 256) {
      float a = accs[d] * es;
#pragma unroll
      for (int i = 0; i < RBLK; ++i) a += pr[i] * Vlds[i * DIM + d];
      accs[d] = a;
    }
    __syncthreads();
  }

  // per-split partial: [m, l, acc[512]]
  float* P = partials + ((size_t)bh * SPLITS + split) * (DIM + 2);
  if (tid == 0) { P[0] = sM; P[1] = sL; }
  P[2 + tid]       = accs[tid];
  P[2 + 256 + tid] = accs[256 + tid];
}

// ---------------------------------------------------------------------------
// Combine the SPLITS partial softmax states per (b,h).
// ---------------------------------------------------------------------------
__global__ void attn_reduce_kernel(const float* __restrict__ partials,
                                   float* __restrict__ out) {
  const int bh  = blockIdx.x;
  const int tid = threadIdx.x;
  const float* base = partials + (size_t)bh * SPLITS * (DIM + 2);
  float M = -1e30f;
  for (int s = 0; s < SPLITS; ++s) M = fmaxf(M, base[s * (DIM + 2)]);
  float w[SPLITS];
  float L = 0.0f;
  for (int s = 0; s < SPLITS; ++s) {
    float e = __expf(base[s * (DIM + 2)] - M);
    w[s] = e;
    L += base[s * (DIM + 2) + 1] * e;
  }
  const float inv = (L > 0.0f) ? (1.0f / L) : 0.0f;
  for (int d = tid; d < DIM; d += 256) {
    float a = 0.0f;
    for (int s = 0; s < SPLITS; ++s) a += w[s] * base[s * (DIM + 2) + 2 + d];
    out[(size_t)bh * DIM + d] = a * inv;
  }
}

// ---------------------------------------------------------------------------
extern "C" void kernel_launch(void* const* d_in, const int* in_sizes, int n_in,
                              void* d_out, int out_size, void* d_ws, size_t ws_size,
                              hipStream_t stream) {
  (void)in_sizes; (void)n_in; (void)out_size; (void)ws_size;
  const float* X    = (const float*)d_in[0];
  const float* Wq_w = (const float*)d_in[1];
  const float* Wq_b = (const float*)d_in[2];
  const float* Wk_w = (const float*)d_in[3];
  const float* Wk_b = (const float*)d_in[4];
  const float* Wv_w = (const float*)d_in[5];
  const float* Wv_b = (const float*)d_in[6];
  float* out = (float*)d_out;

  // workspace layout: Wk_bf16 (512K) | Wv_bf16 (512K) | q (64K) | partials (~1.03M)
  char* ws = (char*)d_ws;
  unsigned short* WkB = (unsigned short*)ws;
  unsigned short* WvB = (unsigned short*)(ws + (512u << 10));
  float* qg           = (float*)(ws + (1024u << 10));
  float* partials     = (float*)(ws + (1024u << 10) + (64u << 10));

  const int n4 = DIM * DIM / 4;  // 65536 float4 per weight matrix
  cvt_bf16_kernel<<<n4 / 256, 256, 0, stream>>>(Wk_w, WkB, n4);
  cvt_bf16_kernel<<<n4 / 256, 256, 0, stream>>>(Wv_w, WvB, n4);
  qproj_kernel<<<32, 256, 0, stream>>>(X, Wq_w, Wq_b, qg);
  attn_kv_kernel<<<32 * SPLITS, 256, 0, stream>>>(X, WkB, WvB, Wk_b, Wv_b, qg,
                                                  partials);
  attn_reduce_kernel<<<32, 256, 0, stream>>>(partials, out);
}
